// GraphAttentionLayer_18949395710218
// MI455X (gfx1250) — matrix operand order repaired and checked by reading, not compiled, not run on hardware
//
#include <hip/hip_runtime.h>
#include <hip/hip_bf16.h>

// ---------------------------------------------------------------------------
// GAT layer, fused for MI455X (gfx1250, wave32, WMMA).
//
// Shapes: B=8, N=2048, F_IN=128, F_OUT=64.
// Cost model: adj read 134MB -> ~5.8us @ 23.3TB/s; transcendentals are the
// compute bound. exp(2(f1_i+f2_j)) factorizes into e1[i]*e2[j], so the inner
// loop needs only 1 v_exp + 1 v_rcp per attention element.
// att matrix never materialized: p-tile built directly in WMMA A-fragment
// registers, accumulated with v_wmma_f32_16x16x32_f16, normalized by row sum
// (tanh bounded in [-1,1] -> no online max), then ELU.
// ---------------------------------------------------------------------------

typedef __attribute__((ext_vector_type(16))) _Float16 v16h;
typedef __attribute__((ext_vector_type(8)))  float    v8f;

#define B_     8
#define N_     2048
#define FIN    128
#define FOUT   64
#define NCHUNK (N_ / 32)      // 64 K-chunks of 32 along the j (softmax) axis

// A-matrix (16-bit, 16x32): group g (0..3) covers VGPRs v=2g,2g+1, i.e. a
// contiguous K run of 4 starting at:
__device__ __forceinline__ int a_gbase(int g, int hw) {
    return (hw << 4) + (g << 2);
}

// ---------------------------------------------------------------------------
// Kernel 0: pack W (128x64 f32) into f16 B-fragment order.
// Wp[kc(4)][ft(4)][lane(32)][v(8)] : u32. 16 waves total.
// ---------------------------------------------------------------------------
__global__ void k_wpack(const float* __restrict__ W, unsigned int* __restrict__ Wp) {
    int tid  = blockIdx.x * blockDim.x + threadIdx.x;
    int lane = tid & 31;
    int w    = tid >> 5;                 // kc*4 + ft
    if (w >= 16) return;
    int ft = w & 3, kc = w >> 2;
    int hw = lane >> 4;
    int col = (ft << 4) + (lane & 15);
    unsigned int vals[8];
#pragma unroll
    for (int v = 0; v < 8; ++v) {
        int k = kc * 32 + (hw << 4) + (v << 1);   // B: lanes16-31 -> K+16
        _Float16 lo = (_Float16)W[(size_t)k * FOUT + col];
        _Float16 hi = (_Float16)W[(size_t)(k + 1) * FOUT + col];
        vals[v] = (unsigned)*(unsigned short*)&lo | ((unsigned)*(unsigned short*)&hi << 16);
    }
    unsigned int* dst = Wp + ((size_t)w * 32 + lane) * 8;
    *(uint4*)(dst)     = make_uint4(vals[0], vals[1], vals[2], vals[3]);
    *(uint4*)(dst + 4) = make_uint4(vals[4], vals[5], vals[6], vals[7]);
}

// ---------------------------------------------------------------------------
// Kernel 1: h = x @ W  (M=16384, K=128, N=64) via WMMA, output f16 row-major.
// One wave per 16-row tile; A loaded as float4 groups, B as packed b128s.
// ---------------------------------------------------------------------------
__global__ void __launch_bounds__(128)
k_gemm_h(const float* __restrict__ x, const unsigned int* __restrict__ Wp,
         _Float16* __restrict__ h16) {
    int tid  = threadIdx.x;
    int lane = tid & 31;
    int wid  = blockIdx.x * 4 + (tid >> 5);   // 1024 waves total
    int b    = wid >> 7;                      // 128 row-tiles per batch
    int i0   = (wid & 127) << 4;
    int hw   = lane >> 4;
    int M    = lane & 15;

    const float* xrow = x + ((size_t)b * N_ + i0 + M) * FIN;

    v8f acc[4] = {};
#pragma unroll
    for (int kc = 0; kc < 4; ++kc) {
        v16h af;
#pragma unroll
        for (int g = 0; g < 4; ++g) {
            float4 xv = *(const float4*)(xrow + kc * 32 + a_gbase(g, hw));
            af[4 * g + 0] = (_Float16)xv.x;
            af[4 * g + 1] = (_Float16)xv.y;
            af[4 * g + 2] = (_Float16)xv.z;
            af[4 * g + 3] = (_Float16)xv.w;
        }
#pragma unroll
        for (int ft = 0; ft < 4; ++ft) {
            const uint4* q = (const uint4*)(Wp + ((size_t)(kc * 4 + ft) * 32 + lane) * 8);
            union { uint4 u[2]; v16h h; } bb;
            bb.u[0] = q[0];
            bb.u[1] = q[1];
            acc[ft] = __builtin_amdgcn_wmma_f32_16x16x32_f16(
                false, af, false, bb.h, (short)0, acc[ft], false, false);
        }
    }
    // C/D layout: vgpr r -> row 8*hw + r, lane%16 -> column
#pragma unroll
    for (int ft = 0; ft < 4; ++ft) {
        int col = (ft << 4) + M;
#pragma unroll
        for (int r = 0; r < 8; ++r) {
            int row = i0 + (hw << 3) + r;
            h16[((size_t)b * N_ + row) * FOUT + col] = (_Float16)acc[ft][r];
        }
    }
}

// ---------------------------------------------------------------------------
// Kernel 2: pack h16 into B-fragment order (L2-resident operand for k_attn).
// hpack[b][chunk][ftile][lane][v] : u32 (two packed f16)
// ---------------------------------------------------------------------------
__global__ void k_pack(const _Float16* __restrict__ h16,
                       unsigned int* __restrict__ hpack) {
    int tid  = blockIdx.x * blockDim.x + threadIdx.x;
    int lane = tid & 31;
    int w    = tid >> 5;                       // (b*64 + c)*4 + ft, 2048 waves
    if (w >= B_ * NCHUNK * 4) return;
    int ft = w & 3;
    int c  = (w >> 2) & (NCHUNK - 1);
    int b  = w >> 8;
    int hw = lane >> 4;
    int col = (ft << 4) + (lane & 15);

    unsigned int vals[8];
#pragma unroll
    for (int v = 0; v < 8; ++v) {
        int k = c * 32 + (hw << 4) + (v << 1);
        unsigned short lo = *(const unsigned short*)(h16 + ((size_t)b * N_ + k)     * FOUT + col);
        unsigned short hi = *(const unsigned short*)(h16 + ((size_t)b * N_ + k + 1) * FOUT + col);
        vals[v] = (unsigned)lo | ((unsigned)hi << 16);
    }
    unsigned int* dst = hpack + ((size_t)w * 32 + lane) * 8;
    *(uint4*)(dst)     = make_uint4(vals[0], vals[1], vals[2], vals[3]);
    *(uint4*)(dst + 4) = make_uint4(vals[4], vals[5], vals[6], vals[7]);
}

// ---------------------------------------------------------------------------
// Kernel 3: e1 = exp(2*h@a[:64]), e2 = exp(2*h@a[64:])  (one thread per row).
// Factorized so the attention inner loop needs one exp, not two.
// ---------------------------------------------------------------------------
__global__ void k_proj(const _Float16* __restrict__ h16,
                       const float* __restrict__ a,
                       float* __restrict__ e1, float* __restrict__ e2) {
    int idx = blockIdx.x * blockDim.x + threadIdx.x;
    if (idx >= B_ * N_) return;
    const _Float16* row = h16 + (size_t)idx * FOUT;
    float s1 = 0.f, s2 = 0.f;
#pragma unroll 8
    for (int f = 0; f < FOUT; ++f) {
        float hv = (float)row[f];
        s1 += hv * a[f];
        s2 += hv * a[FOUT + f];
    }
    e1[idx] = __expf(2.f * s1);
    e2[idx] = __expf(2.f * s2);
}

// ---------------------------------------------------------------------------
// Kernel 4: fused masked-softmax attention + att@h + ELU.
// 4 independent waves per block, 16 rows per wave, j streamed in 32-chunks.
// u = e1[i]*e2[j]; tanh = 1 - 2/(u+1); p = adj ? exp(tanh) : 0.
// ---------------------------------------------------------------------------
__global__ void __launch_bounds__(128)
k_attn(const int* __restrict__ adj, const float* __restrict__ e1,
       const float* __restrict__ e2, const unsigned int* __restrict__ hpack,
       float* __restrict__ out) {
    int tid    = threadIdx.x;
    int lane   = tid & 31;
    int wave   = tid >> 5;
    int b      = blockIdx.x >> 5;              // 32 row-blocks per batch
    int rowblk = blockIdx.x & 31;
    int i0     = rowblk * 64 + wave * 16;
    int hw     = lane >> 4;
    int M      = lane & 15;

    const int*   adjrow = adj + ((size_t)b * N_ + i0 + M) * N_;
    const float  e1m    = e1[(size_t)b * N_ + i0 + M];
    const float* e2b    = e2 + (size_t)b * N_;
    const unsigned int* hpb = hpack + (size_t)b * NCHUNK * 4 * 256;

    v8f   acc[4] = {};
    float rsum   = 0.f;

    for (int c = 0; c < NCHUNK; ++c) {
        int j0 = c << 5;
        if (c + 1 < NCHUNK)                      // gfx1250 global_prefetch_b8
            __builtin_prefetch(adjrow + j0 + 32, 0, 1);

        v16h af;
#pragma unroll
        for (int g = 0; g < 4; ++g) {
            int base = a_gbase(g, hw);
            int4   ad = *(const int4*)(adjrow + j0 + base);
            float4 ev = *(const float4*)(e2b + j0 + base);
            float th0 = 1.f - 2.f * __builtin_amdgcn_rcpf(e1m * ev.x + 1.f);
            float th1 = 1.f - 2.f * __builtin_amdgcn_rcpf(e1m * ev.y + 1.f);
            float th2 = 1.f - 2.f * __builtin_amdgcn_rcpf(e1m * ev.z + 1.f);
            float th3 = 1.f - 2.f * __builtin_amdgcn_rcpf(e1m * ev.w + 1.f);
            float p0 = (ad.x > 0) ? __expf(th0) : 0.f;
            float p1 = (ad.y > 0) ? __expf(th1) : 0.f;
            float p2 = (ad.z > 0) ? __expf(th2) : 0.f;
            float p3 = (ad.w > 0) ? __expf(th3) : 0.f;
            rsum += (p0 + p1) + (p2 + p3);
            af[4 * g + 0] = (_Float16)p0;
            af[4 * g + 1] = (_Float16)p1;
            af[4 * g + 2] = (_Float16)p2;
            af[4 * g + 3] = (_Float16)p3;
        }

        const unsigned int* hp = hpb + (size_t)c * 4 * 256;
#pragma unroll
        for (int ft = 0; ft < 4; ++ft) {
            const uint4* q = (const uint4*)(hp + (size_t)ft * 256 + lane * 8);
            union { uint4 u[2]; v16h h; } bb;
            bb.u[0] = q[0];
            bb.u[1] = q[1];
            acc[ft] = __builtin_amdgcn_wmma_f32_16x16x32_f16(
                false, af, false, bb.h, (short)0, acc[ft], false, false);
        }
    }

    // lanes L and L+16 together cover all K for row L%16
    float total = rsum + __shfl_xor(rsum, 16, 32);
    float inv   = (total > 0.f) ? __builtin_amdgcn_rcpf(total) : 0.f;

#pragma unroll
    for (int ft = 0; ft < 4; ++ft) {
        int col = (ft << 4) + M;
#pragma unroll
        for (int r = 0; r < 8; ++r) {
            int   row = (hw << 3) + r;
            float s   = __shfl(inv, row, 32);   // row sums live on lanes 0..15
            float v   = acc[ft][r] * s;
            v = (v > 0.f) ? v : (__expf(v) - 1.f);   // ELU
            out[((size_t)b * N_ + i0 + row) * FOUT + col] = v;
        }
    }
}

// ---------------------------------------------------------------------------
// Host-side launch sequence (same stream -> serialized dependencies).
// ws layout: h16 (2MB) | hpack (2MB) | e1 (64KB) | e2 (64KB) | Wp (16KB)
// ---------------------------------------------------------------------------
extern "C" void kernel_launch(void* const* d_in, const int* in_sizes, int n_in,
                              void* d_out, int out_size, void* d_ws, size_t ws_size,
                              hipStream_t stream) {
    const float* x   = (const float*)d_in[0];
    const int*   adj = (const int*)  d_in[1];
    const float* W   = (const float*)d_in[2];
    const float* a   = (const float*)d_in[3];
    float*       out = (float*)d_out;

    char* ws = (char*)d_ws;
    _Float16*     h16   = (_Float16*)ws;                                       // 2 MB
    unsigned int* hpack = (unsigned int*)(ws + (size_t)2 * 1024 * 1024);       // 2 MB
    float*        e1    = (float*)(ws + (size_t)4 * 1024 * 1024);              // 64 KB
    float*        e2    = (float*)(ws + (size_t)4 * 1024 * 1024 + 65536);      // 64 KB
    unsigned int* Wp    = (unsigned int*)(ws + (size_t)4 * 1024 * 1024 + 131072); // 16 KB

    // 0) pack W into B-fragment layout: 16 waves
    k_wpack<<<2, 256, 0, stream>>>(W, Wp);
    // 1) h = x @ W (WMMA): 1024 waves
    k_gemm_h<<<256, 128, 0, stream>>>(x, Wp, h16);
    // 2) pack h into B-fragment layout: 2048 waves
    k_pack<<<256, 256, 0, stream>>>(h16, hpack);
    // 3) factorized attention logits: e1/e2
    k_proj<<<(B_ * N_ + 255) / 256, 256, 0, stream>>>(h16, a, e1, e2);
    // 4) fused masked softmax + att@h (WMMA) + ELU: 8 batches x 32 row-blocks
    k_attn<<<B_ * 32, 128, 0, stream>>>(adj, e1, e2, hpack, out);
}
